// TokenSpaceBlock_14963666059917
// MI455X (gfx1250) — matrix operand order repaired and checked by reading, not compiled, not run on hardware
//
#include <hip/hip_runtime.h>
#include <hip/hip_bf16.h>

typedef __bf16 bf16_t;
typedef __attribute__((ext_vector_type(16))) __bf16 v16bf;
typedef __attribute__((ext_vector_type(8)))  float  v8f;

// ---------------- problem constants ----------------
static constexpr int BB    = 4;
static constexpr int NN    = 8192;
static constexpr int CC    = 512;
static constexpr int HH    = 8;
static constexpr int DH    = 64;
static constexpr int GG    = 64;
static constexpr int INNER = HH * DH;      // 512
static constexpr int MROWS = BB * NN;      // 32768

// ---------------- workspace layout (bytes) ----------------
static constexpr size_t OFF_FXN32 = 0;                                   // f32 [MROWS][512]
static constexpr size_t OFF_FXN16 = OFF_FXN32 + (size_t)MROWS*CC*4;      // bf16 [MROWS][512] (reused as slice_w)
static constexpr size_t OFF_XMID  = OFF_FXN16 + (size_t)MROWS*CC*2;      // bf16 [MROWS][512] (reused as out_x)
static constexpr size_t OFF_FXMID = OFF_XMID  + (size_t)MROWS*INNER*2;   // bf16 [MROWS][512]
static constexpr size_t OFF_WX    = OFF_FXMID + (size_t)MROWS*INNER*2;   // bf16 512*512 TRANSPOSED [n][k]
static constexpr size_t OFF_WFX   = OFF_WX    + (size_t)CC*INNER*2;      // transposed
static constexpr size_t OFF_WOUT  = OFF_WFX   + (size_t)CC*INNER*2;      // transposed
static constexpr size_t OFF_WSLT  = OFF_WOUT  + (size_t)INNER*CC*2;      // bf16 64*64 transposed [g][d]
static constexpr size_t OFF_SNORM = OFF_WSLT  + (size_t)DH*GG*2;         // f32 [32][64]
static constexpr size_t OFF_STOK  = OFF_SNORM + (size_t)BB*HH*GG*4;      // f32 [32][64][64]
static constexpr size_t OFF_TOK   = OFF_STOK  + (size_t)BB*HH*GG*DH*4;   // bf16 [32][64][64]

// ---------------- CDNA5 async global->LDS copy (ASYNCcnt) ----------------
__device__ __forceinline__ void async_b128(const bf16_t* gsrc, const bf16_t* ldst) {
  uint32_t l = (uint32_t)(uintptr_t)ldst;   // LDS byte addr = low 32 bits of flat ptr
  uint64_t g = (uint64_t)(uintptr_t)gsrc;
  asm volatile("global_load_async_to_lds_b128 %0, %1, off" :: "v"(l), "v"(g) : "memory");
}
__device__ __forceinline__ void wait_async0() {
  asm volatile("s_wait_asynccnt 0x0" ::: "memory");
}

// ---------------- Tensor Data Mover (TENSORcnt) ----------------
#if defined(__has_builtin)
#if __has_builtin(__builtin_amdgcn_tensor_load_to_lds) && __has_builtin(__builtin_amdgcn_s_wait_tensorcnt)
#define HAVE_TDM 1
#endif
#endif

#ifdef HAVE_TDM
typedef __attribute__((ext_vector_type(4))) unsigned int u32x4;
typedef __attribute__((ext_vector_type(8))) int i32x8;
typedef __attribute__((ext_vector_type(4))) int i32x4;

// 2D bf16 tile load: tile_w elems x tile_h rows, row stride stride_elems, packed into LDS.
// D# per ISA 8.3/8.4: group0 {count=1, lds_addr, global_addr, type=2}; group1 {data_size=1(2B),
// tensor_dim0=stride, tensor_dim1=2^30, tile_dim0/1, tensor_dim0_stride}; trailing groups zero (2D).
__device__ __forceinline__ void tdm_load_2d(const bf16_t* gsrc, const bf16_t* ldst,
                                            unsigned tile_w, unsigned tile_h,
                                            unsigned stride_elems) {
  uint64_t ga = (uint64_t)(uintptr_t)gsrc;
  uint32_t la = (uint32_t)(uintptr_t)ldst;
  const uint32_t td0 = stride_elems, td1 = 1u << 30;
  u32x4 g0 = {};
  g0[0] = 1u;                                            // count=1, user descriptor
  g0[1] = la;                                            // lds_addr
  g0[2] = (uint32_t)ga;                                  // global_addr[31:0]
  g0[3] = (uint32_t)((ga >> 32) & 0x01FFFFFFu) | (2u << 30); // global_addr[56:32] | type=2
  i32x8 g1 = {};
  g1[0] = (int)(1u << 16);                               // data_size = 2 bytes
  g1[1] = (int)((td0 & 0xFFFFu) << 16);                  // tensor_dim0[15:0]
  g1[2] = (int)((td0 >> 16) | ((td1 & 0xFFFFu) << 16));  // tensor_dim0[31:16] | tensor_dim1[15:0]
  g1[3] = (int)((td1 >> 16) | (tile_w << 16));           // tensor_dim1[31:16] | tile_dim0
  g1[4] = (int)(tile_h & 0xFFFFu);                       // tile_dim1 (tile_dim2 = 0)
  g1[5] = (int)stride_elems;                             // tensor_dim0_stride[31:0]
  g1[6] = 0; g1[7] = 0;
  i32x4 z4 = {};
  i32x8 z8 = {};
  __builtin_amdgcn_tensor_load_to_lds(g0, g1, z4, z4, z8, 0);
}
#endif

// ---------------- WMMA fragment helpers (ISA 7.12.2 layouts) ----------------
__device__ __forceinline__ v16bf load_a_frag(const bf16_t* base, int ldk) {
  const int lane = threadIdx.x & 31;
  const int hi = (lane >> 4) & 1, m = lane & 15;
  const bf16_t* p = base + m * ldk + hi * 8;
  union { v16bf v; float4 f[2]; } u;
  u.f[0] = *(const float4*)(p);
  u.f[1] = *(const float4*)(p + 16);
  return u.v;
}
__device__ __forceinline__ v16bf load_b_frag(const bf16_t* baseT, int ldk) {
  const int lane = threadIdx.x & 31;
  const int hi = (lane >> 4) & 1, n = lane & 15;
  const bf16_t* p = baseT + n * ldk + hi * 8;
  union { v16bf v; float4 f[2]; } u;
  u.f[0] = *(const float4*)(p);
  u.f[1] = *(const float4*)(p + 16);
  return u.v;
}
__device__ __forceinline__ v8f wmma_bf16(v16bf a, v16bf b, v8f c) {
  return __builtin_amdgcn_wmma_f32_16x16x32_bf16(false, a, false, b, (short)0, c, false, false);
}

// ---------------- kernel 0: weight conversion (transposed) + zero accumulators ----------------
__global__ __launch_bounds__(256) void k_prep(const float* Wx, const float* Wfx,
    const float* Wout, const float* Wsl,
    bf16_t* WxbT, bf16_t* WfxbT, bf16_t* WoutbT, bf16_t* WslTb,
    float* snorm, float* stok) {
  int i = blockIdx.x * 256 + threadIdx.x;
  if (i < CC * INNER) {
    int k = i >> 9, n = i & 511;
    WxbT [(size_t)n * CC + k]     = (bf16_t)Wx[i];
    WfxbT[(size_t)n * CC + k]     = (bf16_t)Wfx[i];
    WoutbT[(size_t)n * INNER + k] = (bf16_t)Wout[i];
  }
  if (i < DH * GG) { int d = i >> 6, g = i & 63; WslTb[g * 64 + d] = (bf16_t)Wsl[i]; }
  if (i < BB*HH*GG)    snorm[i] = 0.f;
  if (i < BB*HH*GG*DH) stok[i]  = 0.f;
}

// ---------------- kernel 1: LayerNorm (one wave per row) ----------------
__global__ __launch_bounds__(256) void k_ln(const float* fx, const float* g, const float* b,
                                            float* o32, bf16_t* o16) {
  int row  = blockIdx.x * 8 + (threadIdx.x >> 5);
  int lane = threadIdx.x & 31;
  const float* x = fx + (size_t)row * CC + lane * 16;
  float vals[16]; float s = 0.f, ss = 0.f;
#pragma unroll
  for (int i = 0; i < 4; ++i) {
    float4 t = ((const float4*)x)[i];
    vals[4*i+0]=t.x; vals[4*i+1]=t.y; vals[4*i+2]=t.z; vals[4*i+3]=t.w;
    s += t.x+t.y+t.z+t.w; ss += t.x*t.x+t.y*t.y+t.z*t.z+t.w*t.w;
  }
#pragma unroll
  for (int d = 16; d >= 1; d >>= 1) { s += __shfl_xor(s, d, 32); ss += __shfl_xor(ss, d, 32); }
  float mean = s * (1.f/CC);
  float var  = ss * (1.f/CC) - mean*mean;
  float inv  = rsqrtf(var + 1e-5f);
#pragma unroll
  for (int i = 0; i < 16; ++i) {
    int c = lane * 16 + i;
    float y = (vals[i]-mean)*inv*g[c] + b[c];
    o32[(size_t)row*CC + c] = y;
    o16[(size_t)row*CC + c] = (bf16_t)y;
  }
}

// ---------------- kernel 2: dual GEMM, TDM/async double buffered ----------------
__global__ __launch_bounds__(256) void k_gemm_dual(const bf16_t* A, const bf16_t* B1T, const bf16_t* B2T,
    const float* bias1, const float* bias2, bf16_t* O1, bf16_t* O2) {
  __shared__ bf16_t sA [2][128 * 32];
  __shared__ bf16_t sB1[2][64 * 32];
  __shared__ bf16_t sB2[2][64 * 32];
  const int tid = threadIdx.x, wave = tid >> 5, lane = tid & 31;
  const int hi = (lane >> 4) & 1, ln = lane & 15;
  const int row0 = blockIdx.x * 128, col0 = blockIdx.y * 64;
#ifndef HAVE_TDM
  const int ra = tid >> 1,  ca = (tid & 1) * 16;
  const int nb = tid >> 2,  cb = (tid & 3) * 8;
#endif

  auto issue = [&](int k0, int buf) {
#ifdef HAVE_TDM
    if (wave == 0) {
      tdm_load_2d(A   + (size_t)row0 * CC + k0, &sA [buf][0], 32, 128, CC);
      tdm_load_2d(B1T + (size_t)col0 * CC + k0, &sB1[buf][0], 32,  64, CC);
      tdm_load_2d(B2T + (size_t)col0 * CC + k0, &sB2[buf][0], 32,  64, CC);
    }
#else
    const bf16_t* ga = A + (size_t)(row0 + ra) * CC + k0 + ca;
    async_b128(ga,     &sA[buf][ra * 32 + ca]);
    async_b128(ga + 8, &sA[buf][ra * 32 + ca + 8]);
    async_b128(B1T + (size_t)(col0 + nb) * CC + k0 + cb, &sB1[buf][nb * 32 + cb]);
    async_b128(B2T + (size_t)(col0 + nb) * CC + k0 + cb, &sB2[buf][nb * 32 + cb]);
#endif
  };
  auto wait_copies = [&]() {
#ifdef HAVE_TDM
    if (wave == 0) __builtin_amdgcn_s_wait_tensorcnt(0);
#else
    wait_async0();
#endif
    __syncthreads();
  };

  v8f acc1[4] = {}, acc2[4] = {};
  issue(0, 0);
  wait_copies();
  for (int k0 = 0; k0 < CC; k0 += 32) {
    const int buf = (k0 >> 5) & 1;
    if (k0 + 32 < CC) issue(k0 + 32, buf ^ 1);
    v16bf a = load_a_frag(&sA[buf][wave * 16 * 32], 32);
    v16bf fb1[4], fb2[4];
#pragma unroll
    for (int t = 0; t < 4; ++t) {
      fb1[t] = load_b_frag(&sB1[buf][t * 16 * 32], 32);
      fb2[t] = load_b_frag(&sB2[buf][t * 16 * 32], 32);
    }
#pragma unroll
    for (int t = 0; t < 4; ++t) acc1[t] = wmma_bf16(a, fb1[t], acc1[t]);
#pragma unroll
    for (int t = 0; t < 4; ++t) acc2[t] = wmma_bf16(a, fb2[t], acc2[t]);
    wait_copies();
  }
#pragma unroll
  for (int t = 0; t < 4; ++t) {
    int n = col0 + t * 16 + ln;
    float bb1 = bias1[n], bb2 = bias2[n];
#pragma unroll
    for (int r = 0; r < 8; ++r) {
      int m = row0 + wave * 16 + r + hi * 8;
      O1[(size_t)m * INNER + n] = (bf16_t)(acc1[t][r] + bb1);
      O2[(size_t)m * INNER + n] = (bf16_t)(acc2[t][r] + bb2);
    }
  }
}

// ---------------- kernel 3: logits GEMM + softmax + slice_norm ----------------
__global__ __launch_bounds__(256) void k_slice(const bf16_t* xmid, const bf16_t* WslT,
    const float* bslice, const float* temp, bf16_t* w_out, float* snorm) {
  __shared__ bf16_t sA[128 * 64];     // 16 KB
  __shared__ bf16_t sBT[64 * 64];     // 8 KB  (Wslice^T)
  __shared__ float  sLog[8][16 * 64]; // 32 KB
  const int bh = blockIdx.y, b = bh >> 3, h = bh & 7;
  const int n0 = blockIdx.x * 128;
  const int tid = threadIdx.x, wave = tid >> 5, lane = tid & 31;
  const int hi = (lane >> 4) & 1, ln = lane & 15;
#ifdef HAVE_TDM
  if (wave == 0) {
    tdm_load_2d(WslT, sBT, 64, 64, 64);
    tdm_load_2d(xmid + ((size_t)(b * NN + n0)) * INNER + h * DH, sA, 64, 128, INNER);
  }
  if (wave == 0) __builtin_amdgcn_s_wait_tensorcnt(0);
#else
  {
    const bf16_t* gw = WslT + tid * 16;
    async_b128(gw,     sBT + tid * 16);
    async_b128(gw + 8, sBT + tid * 16 + 8);
    int r = tid >> 1, c = (tid & 1) * 32;
    const bf16_t* ga = xmid + ((size_t)(b * NN + n0 + r)) * INNER + h * DH + c;
#pragma unroll
    for (int j = 0; j < 4; ++j)
      async_b128(ga + j * 8, sA + r * 64 + c + j * 8);
  }
  wait_async0();
#endif
  __syncthreads();
  v8f acc[4] = {};
#pragma unroll
  for (int k0 = 0; k0 < 64; k0 += 32) {
    v16bf a = load_a_frag(sA + wave * 16 * 64 + k0, 64);
    v16bf fb[4];
#pragma unroll
    for (int t = 0; t < 4; ++t) fb[t] = load_b_frag(sBT + t * 16 * 64 + k0, 64);
#pragma unroll
    for (int t = 0; t < 4; ++t) acc[t] = wmma_bf16(a, fb[t], acc[t]);
  }
  const float tinv = 1.0f / temp[h];
#pragma unroll
  for (int t = 0; t < 4; ++t) {
    float bs = bslice[t * 16 + ln];
#pragma unroll
    for (int r = 0; r < 8; ++r)
      sLog[wave][(r + hi * 8) * 64 + t * 16 + ln] = (acc[t][r] + bs) * tinv;
  }
  __syncthreads();
  if (lane < 16) { // per-row softmax over G=64
    float* row = &sLog[wave][lane * 64];
    float mx = -1e30f;
    for (int gi = 0; gi < 64; ++gi) mx = fmaxf(mx, row[gi]);
    float s = 0.f;
    for (int gi = 0; gi < 64; ++gi) { float e = __expf(row[gi] - mx); row[gi] = e; s += e; }
    float inv = 1.f / s;
    size_t nrow = (size_t)bh * NN + n0 + wave * 16 + lane;
    for (int gi = 0; gi < 64; ++gi) {
      float wv = row[gi] * inv; row[gi] = wv;
      w_out[nrow * GG + gi] = (bf16_t)wv;
    }
  }
  __syncthreads();
  {
    int c0 = lane * 2;
    float s0 = 0.f, s1 = 0.f;
    for (int r = 0; r < 16; ++r) { s0 += sLog[wave][r * 64 + c0]; s1 += sLog[wave][r * 64 + c0 + 1]; }
    atomicAdd(&snorm[bh * GG + c0],     s0);
    atomicAdd(&snorm[bh * GG + c0 + 1], s1);
  }
}

// ---------------- kernel 4: pooling GEMM ----------------
__global__ __launch_bounds__(256) void k_pool(const bf16_t* w_buf, const bf16_t* fxmid, float* stok) {
  __shared__ bf16_t sWT[64 * 32]; // w^T tile [g][n]
  __shared__ bf16_t sFT[64 * 32]; // fxmid^T tile [d][n]
  const int bh = blockIdx.y, b = bh >> 3, h = bh & 7;
  const int nbase0 = blockIdx.x * 1024;
  const int tid = threadIdx.x, wave = tid >> 5, lane = tid & 31;
  const int hi = (lane >> 4) & 1, ln = lane & 15;
  v8f acc[2] = {};
  for (int it = 0; it < 32; ++it) {
    int nb = nbase0 + it * 32;
    { int nl = tid >> 3, g0 = (tid & 7) * 8;
      const bf16_t* src = w_buf + ((size_t)bh * NN + nb + nl) * GG + g0;
#pragma unroll
      for (int j = 0; j < 8; ++j) sWT[(g0 + j) * 32 + nl] = src[j]; }
    { int nl = tid >> 3, d0 = (tid & 7) * 8;
      const bf16_t* src = fxmid + ((size_t)(b * NN + nb + nl)) * INNER + h * DH + d0;
#pragma unroll
      for (int j = 0; j < 8; ++j) sFT[(d0 + j) * 32 + nl] = src[j]; }
    __syncthreads();
#pragma unroll
    for (int j = 0; j < 2; ++j) {
      int st = wave * 2 + j, gi = st >> 2, di = st & 3;
      v16bf a  = load_a_frag(sWT + gi * 16 * 32, 32);
      v16bf bb = load_b_frag(sFT + di * 16 * 32, 32);
      acc[j] = wmma_bf16(a, bb, acc[j]);
    }
    __syncthreads();
  }
#pragma unroll
  for (int j = 0; j < 2; ++j) {
    int st = wave * 2 + j, gi = st >> 2, di = st & 3;
#pragma unroll
    for (int r = 0; r < 8; ++r)
      atomicAdd(&stok[((size_t)bh * GG + gi * 16 + r + hi * 8) * DH + di * 16 + ln], acc[j][r]);
  }
}

// ---------------- kernel 5: per-(b,h) attention + token MLP (scalar) ----------------
__global__ __launch_bounds__(256) void k_token(const float* stok, const float* snorm,
    const float* Wq, const float* Wk, const float* Wv,
    const float* tg, const float* tb,
    const float* Wm1, const float* bm1, const float* Wm2, const float* bm2,
    bf16_t* tokens) {
  __shared__ float S1[64 * 64], S2[64 * 64], S3[64 * 64], S4[64 * 64];
  const int bh = blockIdx.x, tid = threadIdx.x;
  for (int i = tid; i < 4096; i += 256) {
    int g = i >> 6;
    S1[i] = stok[(size_t)bh * 4096 + i] / (snorm[bh * GG + g] + 1e-5f);
  }
  __syncthreads();
  for (int i = tid; i < 4096; i += 256) {
    int g = i >> 6, d = i & 63; float aq = 0.f, ak = 0.f;
    for (int e = 0; e < 64; ++e) { float x = S1[g * 64 + e]; aq += x * Wq[e * 64 + d]; ak += x * Wk[e * 64 + d]; }
    S2[i] = aq; S3[i] = ak;
  }
  __syncthreads();
  for (int i = tid; i < 4096; i += 256) {
    int g = i >> 6, p = i & 63; float a = 0.f;
    for (int e = 0; e < 64; ++e) a += S2[g * 64 + e] * S3[p * 64 + e];
    S4[i] = a * 0.125f;
  }
  __syncthreads();
  if (tid < 64) {
    float* row = &S4[tid * 64];
    float mx = -1e30f;
    for (int p = 0; p < 64; ++p) mx = fmaxf(mx, row[p]);
    float s = 0.f;
    for (int p = 0; p < 64; ++p) { float e = __expf(row[p] - mx); row[p] = e; s += e; }
    float inv = 1.f / s;
    for (int p = 0; p < 64; ++p) row[p] *= inv;
  }
  __syncthreads();
  for (int i = tid; i < 4096; i += 256) {
    int g = i >> 6, d = i & 63; float a = 0.f;
    for (int e = 0; e < 64; ++e) a += S1[g * 64 + e] * Wv[e * 64 + d];
    S2[i] = a;
  }
  __syncthreads();
  for (int i = tid; i < 4096; i += 256) {
    int g = i >> 6, d = i & 63; float a = 0.f;
    for (int p = 0; p < 64; ++p) a += S4[g * 64 + p] * S2[p * 64 + d];
    S1[i] += a;
  }
  __syncthreads();
  if (tid < 64) {
    float* row = &S1[tid * 64];
    float m = 0.f, vv = 0.f;
    for (int d = 0; d < 64; ++d) m += row[d];
    m *= (1.f / 64.f);
    for (int d = 0; d < 64; ++d) { float x = row[d] - m; vv += x * x; }
    float inv = rsqrtf(vv * (1.f / 64.f) + 1e-5f);
    for (int d = 0; d < 64; ++d) S3[tid * 64 + d] = (row[d] - m) * inv * tg[d] + tb[d];
  }
  __syncthreads();
  for (int i = tid; i < 8192; i += 256) {
    int g = i >> 7, r = i & 127; float a = bm1[r];
    for (int e = 0; e < 64; ++e) a += S3[g * 64 + e] * Wm1[e * 128 + r];
    float ge = 0.5f * a * (1.0f + erff(a * 0.70710678118f));
    if (r < 64) S2[g * 64 + r] = ge; else S4[g * 64 + (r - 64)] = ge;
  }
  __syncthreads();
  for (int i = tid; i < 4096; i += 256) {
    int g = i >> 6, d = i & 63; float a = bm2[d];
    for (int r = 0; r < 64; ++r) a += S2[g * 64 + r] * Wm2[r * 64 + d];
    for (int r = 0; r < 64; ++r) a += S4[g * 64 + r] * Wm2[(64 + r) * 64 + d];
    a += S1[i];
    tokens[(size_t)bh * 4096 + i] = (bf16_t)a;
  }
}

// ---------------- kernel 6: de-slice GEMM ----------------
__global__ __launch_bounds__(256) void k_deslice(const bf16_t* w_buf, const bf16_t* tokens, bf16_t* outx) {
  __shared__ bf16_t sTT[64 * 64]; // tokens^T [d][g]
  const int bh = blockIdx.y, b = bh >> 3, h = bh & 7;
  const int n0 = blockIdx.x * 128;
  const int tid = threadIdx.x, wave = tid >> 5, lane = tid & 31;
  const int hi = (lane >> 4) & 1, ln = lane & 15;
  for (int i = tid; i < 4096; i += 256) { int g = i >> 6, d = i & 63; sTT[d * 64 + g] = tokens[(size_t)bh * 4096 + i]; }
  __syncthreads();
  v8f acc[4] = {};
  const bf16_t* Arow = w_buf + ((size_t)bh * NN + n0 + wave * 16) * GG;
#pragma unroll
  for (int k0 = 0; k0 < 64; k0 += 32) {
    v16bf a = load_a_frag(Arow + k0, 64);
    v16bf fb[4];
#pragma unroll
    for (int t = 0; t < 4; ++t) fb[t] = load_b_frag(sTT + t * 16 * 64 + k0, 64);
#pragma unroll
    for (int t = 0; t < 4; ++t) acc[t] = wmma_bf16(a, fb[t], acc[t]);
  }
#pragma unroll
  for (int t = 0; t < 4; ++t) {
#pragma unroll
    for (int r = 0; r < 8; ++r) {
      int m = n0 + wave * 16 + r + hi * 8;
      int col = h * DH + t * 16 + ln;
      outx[((size_t)b * NN + m) * INNER + col] = (bf16_t)acc[t][r];
    }
  }
}

// ---------------- kernel 7: final GEMM + bias + residual ----------------
__global__ __launch_bounds__(256) void k_out(const bf16_t* A, const bf16_t* BwT,
    const float* bias, const float* resid, float* O) {
  __shared__ bf16_t sA [2][128 * 32];
  __shared__ bf16_t sBT[2][64 * 32];
  const int tid = threadIdx.x, wave = tid >> 5, lane = tid & 31;
  const int hi = (lane >> 4) & 1, ln = lane & 15;
  const int row0 = blockIdx.x * 128, col0 = blockIdx.y * 64;
#ifndef HAVE_TDM
  const int ra = tid >> 1, ca = (tid & 1) * 16;
  const int nb = tid >> 2, cb = (tid & 3) * 8;
#endif

  auto issue = [&](int k0, int buf) {
#ifdef HAVE_TDM
    if (wave == 0) {
      tdm_load_2d(A   + (size_t)row0 * INNER + k0, &sA [buf][0], 32, 128, INNER);
      tdm_load_2d(BwT + (size_t)col0 * INNER + k0, &sBT[buf][0], 32,  64, INNER);
    }
#else
    const bf16_t* ga = A + (size_t)(row0 + ra) * INNER + k0 + ca;
    async_b128(ga,     &sA[buf][ra * 32 + ca]);
    async_b128(ga + 8, &sA[buf][ra * 32 + ca + 8]);
    async_b128(BwT + (size_t)(col0 + nb) * INNER + k0 + cb, &sBT[buf][nb * 32 + cb]);
#endif
  };
  auto wait_copies = [&]() {
#ifdef HAVE_TDM
    if (wave == 0) __builtin_amdgcn_s_wait_tensorcnt(0);
#else
    wait_async0();
#endif
    __syncthreads();
  };

  v8f acc[4] = {};
  issue(0, 0);
  wait_copies();
  for (int k0 = 0; k0 < INNER; k0 += 32) {
    const int buf = (k0 >> 5) & 1;
    if (k0 + 32 < INNER) issue(k0 + 32, buf ^ 1);
    v16bf a = load_a_frag(&sA[buf][wave * 16 * 32], 32);
    v16bf fb[4];
#pragma unroll
    for (int t = 0; t < 4; ++t) fb[t] = load_b_frag(&sBT[buf][t * 16 * 32], 32);
#pragma unroll
    for (int t = 0; t < 4; ++t) acc[t] = wmma_bf16(a, fb[t], acc[t]);
    wait_copies();
  }
#pragma unroll
  for (int t = 0; t < 4; ++t) {
    int n = col0 + t * 16 + ln;
    float bb = bias[n];
#pragma unroll
    for (int r = 0; r < 8; ++r) {
      int m = row0 + wave * 16 + r + hi * 8;
      O[(size_t)m * CC + n] = acc[t][r] + bb + resid[(size_t)m * CC + n];
    }
  }
}

// ---------------- launch ----------------
extern "C" void kernel_launch(void* const* d_in, const int* in_sizes, int n_in,
                              void* d_out, int out_size, void* d_ws, size_t ws_size,
                              hipStream_t stream) {
  const float* fx     = (const float*)d_in[0];
  const float* ln1_g  = (const float*)d_in[1];
  const float* ln1_b  = (const float*)d_in[2];
  const float* Wx     = (const float*)d_in[3];
  const float* bx     = (const float*)d_in[4];
  const float* Wfx    = (const float*)d_in[5];
  const float* bfx    = (const float*)d_in[6];
  const float* Wsl    = (const float*)d_in[7];
  const float* bsl    = (const float*)d_in[8];
  const float* temp   = (const float*)d_in[9];
  const float* Wq     = (const float*)d_in[10];
  const float* Wk     = (const float*)d_in[11];
  const float* Wv     = (const float*)d_in[12];
  const float* Wout   = (const float*)d_in[13];
  const float* bout   = (const float*)d_in[14];
  const float* tln_g  = (const float*)d_in[15];
  const float* tln_b  = (const float*)d_in[16];
  const float* Wm1    = (const float*)d_in[17];
  const float* bm1    = (const float*)d_in[18];
  const float* Wm2    = (const float*)d_in[19];
  const float* bm2    = (const float*)d_in[20];
  float* out = (float*)d_out;

  unsigned char* W = (unsigned char*)d_ws;
  float*  fxn32  = (float*) (W + OFF_FXN32);
  bf16_t* fxn16  = (bf16_t*)(W + OFF_FXN16);
  bf16_t* w_bf   = fxn16;
  bf16_t* xmid   = (bf16_t*)(W + OFF_XMID);
  bf16_t* outx   = xmid;
  bf16_t* fxmid  = (bf16_t*)(W + OFF_FXMID);
  bf16_t* WxbT   = (bf16_t*)(W + OFF_WX);
  bf16_t* WfxbT  = (bf16_t*)(W + OFF_WFX);
  bf16_t* WoutbT = (bf16_t*)(W + OFF_WOUT);
  bf16_t* WslTb  = (bf16_t*)(W + OFF_WSLT);
  float*  snorm  = (float*) (W + OFF_SNORM);
  float*  stok   = (float*) (W + OFF_STOK);
  bf16_t* tokens = (bf16_t*)(W + OFF_TOK);

  k_prep<<<dim3(1024), dim3(256), 0, stream>>>(Wx, Wfx, Wout, Wsl,
      WxbT, WfxbT, WoutbT, WslTb, snorm, stok);
  k_ln<<<dim3(MROWS / 8), dim3(256), 0, stream>>>(fx, ln1_g, ln1_b, fxn32, fxn16);
  k_gemm_dual<<<dim3(MROWS / 128, INNER / 64), dim3(256), 0, stream>>>(
      fxn16, WxbT, WfxbT, bx, bfx, xmid, fxmid);
  k_slice<<<dim3(NN / 128, BB * HH), dim3(256), 0, stream>>>(
      xmid, WslTb, bsl, temp, w_bf, snorm);
  k_pool<<<dim3(NN / 1024, BB * HH), dim3(256), 0, stream>>>(w_bf, fxmid, stok);
  k_token<<<dim3(BB * HH), dim3(256), 0, stream>>>(stok, snorm, Wq, Wk, Wv,
      tln_g, tln_b, Wm1, bm1, Wm2, bm2, tokens);
  k_deslice<<<dim3(NN / 128, BB * HH), dim3(256), 0, stream>>>(w_bf, tokens, outx);
  k_out<<<dim3(MROWS / 128, CC / 64), dim3(256), 0, stream>>>(outx, WoutbT, bout, fxn32, out);
}
